// channelMix_CI_pat_50311246905750
// MI455X (gfx1250) — compile-verified, matrix operand order checked by reading, not run on hardware
//
#include <hip/hip_runtime.h>
#include <hip/hip_bf16.h>
#include <math.h>

// ---------------------------------------------------------------------------
// MI455X channel-mixing MLP:
//   K0: convert W1/W2/b1/b2 f32->f16 into scratch (enables async LDS staging)
//   K1: per-channel [32x64]x[64x64] WMMA -> exact GELU -> WMMA -> +x (f16 res)
//   K2: LayerNorm over C
// B=1024, P=64, C=512. v_wmma_f32_16x16x32_f16, f32 accumulate.
// Weights staged into LDS via global_load_async_to_lds_b128 (ASYNCcnt).
// ---------------------------------------------------------------------------

typedef _Float16 half_t;
typedef __attribute__((ext_vector_type(16))) _Float16 v16h;
typedef __attribute__((ext_vector_type(8)))  _Float16 v8h;
typedef __attribute__((ext_vector_type(4)))  _Float16 v4h;
typedef __attribute__((ext_vector_type(8)))  float    v8f;

static constexpr int Bsz = 1024;   // batch
static constexpr int Pp  = 64;     // patches
static constexpr int Cc  = 512;    // channels
static constexpr int CG  = 16;     // channels per workgroup (64B-coalesced x loads)
static constexpr int BT  = 32;     // batch rows per workgroup
static constexpr int ROW = 72;     // 64 + 8 halves pad: 144B rows, 16B aligned

// LDS offsets (in halves)
static constexpr int XT_OFF  = 0;                       // x tile  [CG][BT][ROW]
static constexpr int W1_OFF  = XT_OFF + CG * BT * ROW;  // W1[c]   [64][ROW]  (B-matrix [n][k])
static constexpr int W2_OFF  = W1_OFF + 64 * ROW;       // W2[c]
static constexpr int HT_OFF  = W2_OFF + 64 * ROW;       // GELU(h) [BT][ROW] handoff
static constexpr int B1_OFF  = HT_OFF + BT * ROW;       // bias1 [64]
static constexpr int B2_OFF  = B1_OFF + 64;             // bias2 [64]
static constexpr int RES_OFF = B2_OFF + 64;             // res tile [BT][64][CG] f16
static constexpr int LDS_HALVES = RES_OFF + BT * 64 * CG;   // 81,280 halves
static constexpr int LDS_BYTES  = LDS_HALVES * 2;           // 162,560 B (2 WGs / 320KB WGP)

// WMMA fragment from an LDS row per ISA layout:
// halves 0..7 = K[kk + 8*khi .. +7], halves 8..15 = K[kk + 16 + 8*khi .. +7]
__device__ __forceinline__ v16h load_frag(const half_t* row, int kk, int khi) {
    v8h lo = *(const v8h*)(row + kk + 8 * khi);        // ds_load_b128
    v8h hi = *(const v8h*)(row + kk + 16 + 8 * khi);   // ds_load_b128
    v16h r;
#pragma unroll
    for (int i = 0; i < 8; ++i) { r[i] = lo[i]; r[i + 8] = hi[i]; }
    return r;
}

// ---- K0: bulk f32 -> f16 conversion (vectorized), for weights & biases
__global__ __launch_bounds__(256)
void cvt_f32_to_f16_k0(const float* __restrict__ src, half_t* __restrict__ dst, int n4) {
    const int i = blockIdx.x * 256 + threadIdx.x;
    if (i < n4) {
        const float4 v = ((const float4*)src)[i];
        ((v4h*)dst)[i] = (v4h){(half_t)v.x, (half_t)v.y, (half_t)v.z, (half_t)v.w};
    }
}

__global__ __launch_bounds__(256)
void mixer_gemm_k1(const float* __restrict__ x,
                   const half_t* __restrict__ W1h, const half_t* __restrict__ b1h,
                   const half_t* __restrict__ W2h, const half_t* __restrict__ b2h,
                   half_t* __restrict__ resh /* [B][P][C] f16 */) {
    extern __shared__ half_t smem[];
    const int tid  = threadIdx.x;
    const int c0   = blockIdx.x * CG;   // channel group base
    const int b0   = blockIdx.y * BT;   // batch tile base
    const int lane = tid & 31;
    const int wave = tid >> 5;
    const int lm   = lane & 15;
    const int khi  = lane >> 4;
    const int m0   = (wave & 1) * 16;   // 2 M-groups
    const int n0   = (wave >> 1) * 16;  // 4 N-groups
    // Dynamic LDS starts after static LDS (0 here); hedge with groupstaticsize.
    const unsigned lds_base = __builtin_amdgcn_groupstaticsize();

    // ---- Phase 0: coalesced x load (64B lines), f32->f16 transpose into LDS
    for (int it = tid; it < BT * 64 * 4; it += 256) {
        const int chunk = it & 3;            // 4-float sub-chunk of the 16 channels
        const int p     = (it >> 2) & 63;
        const int m     = it >> 8;           // 0..31
        const float4 v  = *(const float4*)(x + ((size_t)(b0 + m) * Pp + p) * Cc + c0 + chunk * 4);
        const float vv[4] = {v.x, v.y, v.z, v.w};
#pragma unroll
        for (int j = 0; j < 4; ++j) {
            const int cl = chunk * 4 + j;
            smem[XT_OFF + (cl * BT + m) * ROW + p] = (half_t)vv[j];
        }
    }

    // ---- Channel loop
    for (int ci = 0; ci < CG; ++ci) {
        const int c = c0 + ci;
        __syncthreads();   // prior readers of W/HT done; XT ready (ci==0)

        // Stage weights into padded LDS rows via async B128 copies (ASYNCcnt).
        // Per matrix: 64 rows x 8 x 16B = 512 transfers; 256 threads x 2 iters.
        for (int it = tid; it < 512; it += 256) {
            const int q = it >> 3, seg = it & 7;
            const unsigned lds1 = lds_base + (unsigned)((W1_OFF + q * ROW + seg * 8) * 2);
            const unsigned lds2 = lds_base + (unsigned)((W2_OFF + q * ROW + seg * 8) * 2);
            const unsigned long long ga1 =
                (unsigned long long)(uintptr_t)(W1h + (size_t)c * 4096 + q * 64 + seg * 8);
            const unsigned long long ga2 =
                (unsigned long long)(uintptr_t)(W2h + (size_t)c * 4096 + q * 64 + seg * 8);
            asm volatile("global_load_async_to_lds_b128 %0, %1, off"
                         :: "v"(lds1), "v"(ga1) : "memory");
            asm volatile("global_load_async_to_lds_b128 %0, %1, off"
                         :: "v"(lds2), "v"(ga2) : "memory");
        }
        if (tid < 64)       smem[B1_OFF + tid]      = b1h[(size_t)c * 64 + tid];
        else if (tid < 128) smem[B2_OFF + tid - 64] = b2h[(size_t)c * 64 + (tid - 64)];
        asm volatile("s_wait_asynccnt 0x0" ::: "memory");
        __syncthreads();

        // Prefetch next channel's f16 weights (global_prefetch_b8), 8KB each
        if (ci + 1 < CG) {
            if (tid < 128)
                __builtin_prefetch((const char*)(W1h + (size_t)(c + 1) * 4096) + tid * 64, 0, 0);
            else
                __builtin_prefetch((const char*)(W2h + (size_t)(c + 1) * 4096) + (tid - 128) * 64, 0, 0);
        }

        // ---- Stage 1: H = X * W1^T + b1, exact GELU, handoff via LDS
        {
            const half_t* arow = smem + XT_OFF + (ci * BT + m0 + lm) * ROW;
            const half_t* brow = smem + W1_OFF + (n0 + lm) * ROW;
            const float bias1 = (float)smem[B1_OFF + n0 + lm];
            v8f acc;
#pragma unroll
            for (int r = 0; r < 8; ++r) acc[r] = bias1;
#pragma unroll
            for (int ks = 0; ks < 2; ++ks) {
                const v16h a = load_frag(arow, ks * 32, khi);
                const v16h b = load_frag(brow, ks * 32, khi);
                acc = __builtin_amdgcn_wmma_f32_16x16x32_f16(false, a, false, b,
                                                             (short)0, acc, false, false);
            }
#pragma unroll
            for (int r = 0; r < 8; ++r) {
                const float v = acc[r];
                const float g = 0.5f * v * (1.0f + erff(v * 0.70710678118654752f)); // exact GELU
                smem[HT_OFF + (m0 + r + 8 * khi) * ROW + n0 + lm] = (half_t)g;
            }
        }
        __syncthreads();

        // ---- Stage 2: O = H * W2^T + b2; res = O + x; store f16 res tile
        {
            const half_t* arow = smem + HT_OFF + (m0 + lm) * ROW;
            const half_t* brow = smem + W2_OFF + (n0 + lm) * ROW;
            const float bias2 = (float)smem[B2_OFF + n0 + lm];
            v8f acc;
#pragma unroll
            for (int r = 0; r < 8; ++r) acc[r] = bias2;
#pragma unroll
            for (int ks = 0; ks < 2; ++ks) {
                const v16h a = load_frag(arow, ks * 32, khi);
                const v16h b = load_frag(brow, ks * 32, khi);
                acc = __builtin_amdgcn_wmma_f32_16x16x32_f16(false, a, false, b,
                                                             (short)0, acc, false, false);
            }
            // Gather residual x first (pipelined ds_load_u16, single wait) ...
            float xv[8];
#pragma unroll
            for (int r = 0; r < 8; ++r)
                xv[r] = (float)smem[XT_OFF + (ci * BT + m0 + r + 8 * khi) * ROW + n0 + lm];
            // ... then add + scatter into the [m][q][c] res tile.
#pragma unroll
            for (int r = 0; r < 8; ++r)
                smem[RES_OFF + ((m0 + r + 8 * khi) * 64 + n0 + lm) * CG + ci] =
                    (half_t)(acc[r] + xv[r]);
        }
    }
    __syncthreads();

    // ---- Writeout: res tile -> scratch [B][P][C] f16, 16B vector stores
    for (int it = tid; it < BT * 64 * 2; it += 256) {
        const int chunk = it & 1;
        const int q     = (it >> 1) & 63;
        const int m     = it >> 7;
        const v8h hv = *(const v8h*)(smem + RES_OFF + (m * 64 + q) * CG + chunk * 8);
        *(v8h*)(resh + ((size_t)(b0 + m) * Pp + q) * Cc + c0 + chunk * 8) = hv;
    }
}

// ---- K2: LayerNorm over C. One wave32 per (b,q) row (1KB contiguous f16).
__global__ __launch_bounds__(256)
void mixer_ln_k2(const half_t* __restrict__ resh,
                 const float* __restrict__ gamma, const float* __restrict__ beta,
                 float* __restrict__ out) {
    const int lane = threadIdx.x & 31;
    const int wave = threadIdx.x >> 5;
    const size_t row = (size_t)blockIdx.x * 8 + wave;   // B*P = 65536 rows

    const half_t* rp = resh + row * Cc + lane * 16;
    const v8h a = *(const v8h*)rp;
    const v8h b = *(const v8h*)(rp + 8);
    float v[16], s = 0.f, s2 = 0.f;
#pragma unroll
    for (int i = 0; i < 8; ++i) { v[i] = (float)a[i]; v[i + 8] = (float)b[i]; }
#pragma unroll
    for (int i = 0; i < 16; ++i) { s += v[i]; s2 += v[i] * v[i]; }
#pragma unroll
    for (int off = 16; off > 0; off >>= 1) {
        s  += __shfl_xor(s,  off, 32);
        s2 += __shfl_xor(s2, off, 32);
    }
    const float mean = s * (1.0f / Cc);
    const float var  = s2 * (1.0f / Cc) - mean * mean;
    const float rstd = rsqrtf(var + 1e-5f);

    const int cb = lane * 16;
#pragma unroll
    for (int seg = 0; seg < 4; ++seg) {
        float4 o;
        o.x = (v[seg * 4 + 0] - mean) * rstd * gamma[cb + seg * 4 + 0] + beta[cb + seg * 4 + 0];
        o.y = (v[seg * 4 + 1] - mean) * rstd * gamma[cb + seg * 4 + 1] + beta[cb + seg * 4 + 1];
        o.z = (v[seg * 4 + 2] - mean) * rstd * gamma[cb + seg * 4 + 2] + beta[cb + seg * 4 + 2];
        o.w = (v[seg * 4 + 3] - mean) * rstd * gamma[cb + seg * 4 + 3] + beta[cb + seg * 4 + 3];
        *(float4*)(out + row * Cc + cb + seg * 4) = o;
    }
}

extern "C" void kernel_launch(void* const* d_in, const int* in_sizes, int n_in,
                              void* d_out, int out_size, void* d_ws, size_t ws_size,
                              hipStream_t stream) {
    const float* x     = (const float*)d_in[0];
    const float* W1    = (const float*)d_in[1];
    const float* b1    = (const float*)d_in[2];
    const float* W2    = (const float*)d_in[3];
    const float* b2    = (const float*)d_in[4];
    const float* gamma = (const float*)d_in[5];
    const float* beta  = (const float*)d_in[6];
    float* out = (float*)d_out;

    // Scratch layout (f16): res[B*P*C] | W1h | W2h | b1h | b2h  (~72.2 MB)
    half_t* resh = (half_t*)d_ws;
    half_t* W1h  = resh + (size_t)Bsz * Pp * Cc;
    half_t* W2h  = W1h + (size_t)Cc * Pp * Pp;
    half_t* b1h  = W2h + (size_t)Cc * Pp * Pp;
    half_t* b2h  = b1h + (size_t)Cc * Pp;

    // K0: f32 -> f16 weight/bias conversion
    const int w4 = Cc * Pp * Pp / 4;   // 524288 float4
    const int v4 = Cc * Pp / 4;        // 8192  float4
    cvt_f32_to_f16_k0<<<(w4 + 255) / 256, 256, 0, stream>>>(W1, W1h, w4);
    cvt_f32_to_f16_k0<<<(w4 + 255) / 256, 256, 0, stream>>>(W2, W2h, w4);
    cvt_f32_to_f16_k0<<<(v4 + 255) / 256, 256, 0, stream>>>(b1, b1h, v4);
    cvt_f32_to_f16_k0<<<(v4 + 255) / 256, 256, 0, stream>>>(b2, b2h, v4);

    // K1: fused GEMM/GELU/GEMM/residual per channel group
    dim3 g1(Cc / CG, Bsz / BT);     // 32 x 32 workgroups
    mixer_gemm_k1<<<g1, 256, LDS_BYTES, stream>>>(x, W1h, b1h, W2h, b2h, resh);

    // K2: LayerNorm
    dim3 g2((Bsz * Pp) / 8);        // 8192 workgroups, 8 rows each
    mixer_ln_k2<<<g2, 256, 0, stream>>>(resh, gamma, beta, out);
}
